// PredicateCLSModel_88210038325680
// MI455X (gfx1250) — matrix operand order repaired and checked by reading, not compiled, not run on hardware
//
#include <hip/hip_runtime.h>
#include <cstdint>

// ---------------------------------------------------------------------------
// MI455X (gfx1250) PredicateCLSModel forward loss.
//   - All dense / gathered GEMMs on v_wmma_f32_16x16x32_bf16 (fp32 accum).
//   - Activations & weights pre-converted to bf16 (weights pre-transposed to
//     K-contiguous) so every WMMA fragment is two b128 loads per lane:
//     zero conversions / zero LDS staging in the hot loops.
//   - embed/m (25.6MB bf16) are L2-resident (192MB L2); scatter uses hardware
//     global_atomic_add_f32 into an fp32 accumulator.
// ---------------------------------------------------------------------------

typedef __bf16 bf16_t;
typedef bf16_t v4bf  __attribute__((ext_vector_type(4)));
typedef bf16_t v8bf  __attribute__((ext_vector_type(8)));
typedef bf16_t v16bf __attribute__((ext_vector_type(16)));
typedef float  v8f   __attribute__((ext_vector_type(8)));

#define CAT16(lo, hi) \
  __builtin_shufflevector(lo, hi, 0, 1, 2, 3, 4, 5, 6, 7, 8, 9, 10, 11, 12, 13, 14, 15)

__device__ __forceinline__ float bce_term(float z, float y) {
  return fmaxf(z, 0.f) - z * y + log1pf(expf(-fabsf(z)));
}

__device__ __forceinline__ void block_reduce_add2(float a, float b,
                                                  float* dst_a, float* dst_b,
                                                  float* red) {
  const int tid = threadIdx.x;
  red[tid] = a;
  __syncthreads();
  for (int s = blockDim.x >> 1; s > 0; s >>= 1) {
    if (tid < s) red[tid] += red[tid + s];
    __syncthreads();
  }
  if (tid == 0) atomicAdd(dst_a, red[0]);
  __syncthreads();
  red[tid] = b;
  __syncthreads();
  for (int s = blockDim.x >> 1; s > 0; s >>= 1) {
    if (tid < s) red[tid] += red[tid + s];
    __syncthreads();
  }
  if (tid == 0) atomicAdd(dst_b, red[0]);
}

// ---------------------------------------------------------------------------
// setup kernels
// ---------------------------------------------------------------------------
__global__ void k_zero_acc(float* acc) {
  if (threadIdx.x < 16) acc[threadIdx.x] = 0.f;
}

__global__ void k_init_deg(int* deg, int N) {
  int i = blockIdx.x * blockDim.x + threadIdx.x;
  if (i < N) deg[i] = 1;  // self loop
}

__global__ void k_count_deg(const int* __restrict__ trg, int* __restrict__ deg, int E) {
  int e = blockIdx.x * blockDim.x + threadIdx.x;
  if (e < E) atomicAdd(&deg[trg[e]], 1);
}

__global__ void k_dis(const int* __restrict__ deg, float* __restrict__ dis, int N) {
  int i = blockIdx.x * blockDim.x + threadIdx.x;
  if (i < N) dis[i] = rsqrtf((float)deg[i]);
}

// W [K x Nc] f32 row-major -> Wt [Nc x K] bf16 (K-contiguous)
__global__ void k_transpose_w(const float* __restrict__ W, bf16_t* __restrict__ Wt,
                              int K, int Nc) {
  int idx = blockIdx.x * blockDim.x + threadIdx.x;
  if (idx >= K * Nc) return;
  int k = idx / Nc, n = idx % Nc;
  Wt[(size_t)n * K + k] = (bf16_t)W[idx];
}

// fused edge-head weight: Bt[c][k], c in [0,64): 0..18 rel, 19..51 ta, rest 0
__global__ void k_build_wcat(const float* __restrict__ W_rel,
                             const float* __restrict__ W_ta,
                             bf16_t* __restrict__ Bt, int K2) {
  int idx = blockIdx.x * blockDim.x + threadIdx.x;
  if (idx >= 64 * K2) return;
  int c = idx / K2, k = idx % K2;
  float v = 0.f;
  if (c < 19)      v = W_rel[(size_t)k * 19 + c];
  else if (c < 52) v = W_ta[(size_t)k * 33 + (c - 19)];
  Bt[idx] = (bf16_t)v;
}

__global__ __launch_bounds__(256)
void k_f32_to_bf16(const float* __restrict__ in, bf16_t* __restrict__ out, size_t n4) {
  size_t i = (size_t)blockIdx.x * blockDim.x + threadIdx.x;
  if (i >= n4) return;
  float4 v = ((const float4*)in)[i];
  v4bf o;
  o[0] = (bf16_t)v.x; o[1] = (bf16_t)v.y; o[2] = (bf16_t)v.z; o[3] = (bf16_t)v.w;
  ((v4bf*)out)[i] = o;
}

// ---------------------------------------------------------------------------
// Dense GEMM: C[M,Nc] = bf16( (A[M,K] @ B) (+bias) (ReLU) )
// A bf16 row-major, Bt bf16 [Nc][K] (K-contiguous). Block = 8 waves, each
// wave one 16x16 tile; block tile = 16 rows x 128 cols. Fragments are pure
// b128 loads per the ISA register layouts; no LDS, no conversions.
// ---------------------------------------------------------------------------
__global__ __launch_bounds__(256)
void k_gemm_wmma(const bf16_t* __restrict__ A, const bf16_t* __restrict__ Bt,
                 const float* __restrict__ bias, bf16_t* __restrict__ C,
                 int M, int K, int Nc, int do_relu) {
  const int tid  = threadIdx.x;
  const int lane = tid & 31;
  const int wave = tid >> 5;
  const int hl   = lane >> 4;  // lane half selects K sub-range
  const int n    = lane & 15;
  const int row0 = blockIdx.x * 16;
  const int col0 = blockIdx.y * 128 + wave * 16;

  const int rowA = min(row0 + n, M - 1);  // clamp; invalid rows masked at store
  const bf16_t* __restrict__ Arow = A + (size_t)rowA * K + hl * 8;
  const bf16_t* __restrict__ Brow = Bt + (size_t)(col0 + n) * K + hl * 8;

  v8f acc = {};
#pragma unroll 2
  for (int k0 = 0; k0 < K; k0 += 32) {
    // A frag: e 0..7 -> K = k0+hl*8+e ; e 8..15 -> K = k0+hl*8+16+(e-8)
    v8bf alo = *(const v8bf*)(Arow + k0);
    v8bf ahi = *(const v8bf*)(Arow + k0 + 16);
    v16bf af = CAT16(alo, ahi);
    // B frag: e 0..7 -> K = k0+hl*8+e ; e 8..15 -> K = k0+hl*8+16+(e-8)
    v8bf blo = *(const v8bf*)(Brow + k0);
    v8bf bhi = *(const v8bf*)(Brow + k0 + 16);
    v16bf bfv = CAT16(blo, bhi);
    acc = __builtin_amdgcn_wmma_f32_16x16x32_bf16(false, af, false, bfv,
                                                  (short)0, acc, false, false);
  }

  const int col = col0 + n;
  const float bb = bias ? bias[col] : 0.f;
#pragma unroll
  for (int r = 0; r < 8; ++r) {
    int row = row0 + r + 8 * hl;
    if (row < M) {
      float v = acc[r] + bb;
      if (do_relu) v = fmaxf(v, 0.f);
      C[(size_t)row * Nc + col] = (bf16_t)v;
    }
  }
}

// ---------------------------------------------------------------------------
// agg = m * snorm  (snorm = dis^2 = 1/deg), m is bf16
// ---------------------------------------------------------------------------
__global__ __launch_bounds__(256)
void k_scale_init(const bf16_t* __restrict__ m, const float* __restrict__ dis,
                  float* __restrict__ agg, int N, int D) {
  size_t idx = (size_t)blockIdx.x * blockDim.x + threadIdx.x;
  size_t total = (size_t)N * D / 4;
  if (idx >= total) return;
  int node = (int)((idx * 4) / D);
  float w = dis[node];
  w *= w;
  v4bf v = ((const v4bf*)m)[idx];
  float4 o;
  o.x = (float)v[0] * w; o.y = (float)v[1] * w;
  o.z = (float)v[2] * w; o.w = (float)v[3] * w;
  ((float4*)agg)[idx] = o;
}

// ---------------------------------------------------------------------------
// scatter: agg[trg] += m[src] * (dis[src]*dis[trg]); block per edge,
// hardware f32 atomics into L2-resident agg.
// ---------------------------------------------------------------------------
__global__ __launch_bounds__(128)
void k_scatter(const bf16_t* __restrict__ m, const int* __restrict__ src,
               const int* __restrict__ trg, const float* __restrict__ dis,
               float* __restrict__ agg, int E, int D) {
  int e = blockIdx.x;
  if (e >= E) return;
  int s = src[e], t = trg[e];
  float w = dis[s] * dis[t];
  const v4bf* mm = (const v4bf*)(m + (size_t)s * D);
  float* ag = agg + (size_t)t * D;
  int d = threadIdx.x;
  v4bf v = mm[d];
  unsafeAtomicAdd(&ag[d * 4 + 0], (float)v[0] * w);
  unsafeAtomicAdd(&ag[d * 4 + 1], (float)v[1] * w);
  unsafeAtomicAdd(&ag[d * 4 + 2], (float)v[2] * w);
  unsafeAtomicAdd(&ag[d * 4 + 3], (float)v[3] * w);
}

__global__ __launch_bounds__(256)
void k_bias_relu(const float* __restrict__ agg, const float* __restrict__ bias,
                 bf16_t* __restrict__ out, int N, int D) {
  size_t idx = (size_t)blockIdx.x * blockDim.x + threadIdx.x;
  size_t total = (size_t)N * D / 4;
  if (idx >= total) return;
  int dcol = (int)((idx * 4) % D);
  float4 v = ((const float4*)agg)[idx];
  float4 b = *(const float4*)(bias + dcol);
  v4bf o;
  o[0] = (bf16_t)fmaxf(v.x + b.x, 0.f);
  o[1] = (bf16_t)fmaxf(v.y + b.y, 0.f);
  o[2] = (bf16_t)fmaxf(v.z + b.z, 0.f);
  o[3] = (bf16_t)fmaxf(v.w + b.w, 0.f);
  ((v4bf*)out)[idx] = o;
}

// ---------------------------------------------------------------------------
// node heads: thread per (node, head) over 13 heads (4 att + 9 ia)
// ---------------------------------------------------------------------------
__global__ __launch_bounds__(256)
void k_node_heads(const bf16_t* __restrict__ embed,
                  const float* __restrict__ W_att, const float* __restrict__ b_att,
                  const float* __restrict__ W_ia,  const float* __restrict__ b_ia,
                  const float* __restrict__ node_att, const float* __restrict__ node_ia,
                  float* __restrict__ acc, int N, int D) {
  __shared__ float red[256];
  int gid = blockIdx.x * blockDim.x + threadIdx.x;
  int node = gid / 13, head = gid % 13;
  float att_s = 0.f, ia_s = 0.f;
  if (node < N) {
    const bf16_t* row = embed + (size_t)node * D;
    if (head < 4) {
      float z = b_att[head];
      for (int k = 0; k < D; ++k) z += (float)row[k] * W_att[k * 4 + head];
      att_s = bce_term(z, node_att[(size_t)node * 4 + head]);
    } else {
      int h = head - 4;
      float z = b_ia[h];
      for (int k = 0; k < D; ++k) z += (float)row[k] * W_ia[k * 9 + h];
      ia_s = bce_term(z, node_ia[(size_t)node * 9 + h]);
    }
  }
  block_reduce_add2(att_s, ia_s, &acc[0], &acc[1], red);
}

// ---------------------------------------------------------------------------
// edge heads: gather-WMMA GEMM. Wave tile = 16 edges x 64 cols (52 valid).
// K = 2D: k<D from embed[src], else embed[trg] (bf16, L2-resident).
// Bt = fused [64][2D] bf16 K-contiguous -> all fragments are b128 loads.
// BCE in epilogue, block-reduced into global accumulators.
// ---------------------------------------------------------------------------
__global__ __launch_bounds__(256)
void k_edge_heads(const bf16_t* __restrict__ embed,
                  const int* __restrict__ src, const int* __restrict__ trg,
                  const bf16_t* __restrict__ Bt,
                  const float* __restrict__ b_rel, const float* __restrict__ b_ta,
                  const float* __restrict__ edge_rel, const float* __restrict__ edge_ta,
                  float* __restrict__ acc, int E, int D) {
  __shared__ float red[256];
  const int tid  = threadIdx.x;
  const int lane = tid & 31;
  const int wave = tid >> 5;
  const int hl   = lane >> 4;
  const int n    = lane & 15;
  const int tile  = blockIdx.x * 8 + wave;
  const int ebase = tile * 16;
  const int ecl = min(ebase + n, E - 1);
  const int s = src[ecl];
  const int t = trg[ecl];
  const int K2 = 2 * D;

  const bf16_t* __restrict__ As = embed + (size_t)s * D + hl * 8;
  const bf16_t* __restrict__ At = embed + (size_t)t * D + hl * 8;
  const bf16_t* __restrict__ B0 = Bt + (size_t)(n)      * K2 + hl * 8;
  const bf16_t* __restrict__ B1 = Bt + (size_t)(16 + n) * K2 + hl * 8;
  const bf16_t* __restrict__ B2 = Bt + (size_t)(32 + n) * K2 + hl * 8;
  const bf16_t* __restrict__ B3 = Bt + (size_t)(48 + n) * K2 + hl * 8;

  v8f acc0 = {}, acc1 = {}, acc2 = {}, acc3 = {};
#pragma unroll 2
  for (int k0 = 0; k0 < K2; k0 += 32) {
    const bf16_t* ab = (k0 < D) ? (As + k0) : (At + (k0 - D));
    v8bf alo = *(const v8bf*)(ab);
    v8bf ahi = *(const v8bf*)(ab + 16);
    v16bf af = CAT16(alo, ahi);
    v16bf b0 = CAT16(*(const v8bf*)(B0 + k0), *(const v8bf*)(B0 + k0 + 16));
    v16bf b1 = CAT16(*(const v8bf*)(B1 + k0), *(const v8bf*)(B1 + k0 + 16));
    v16bf b2 = CAT16(*(const v8bf*)(B2 + k0), *(const v8bf*)(B2 + k0 + 16));
    v16bf b3 = CAT16(*(const v8bf*)(B3 + k0), *(const v8bf*)(B3 + k0 + 16));
    acc0 = __builtin_amdgcn_wmma_f32_16x16x32_bf16(false, af, false, b0, (short)0, acc0, false, false);
    acc1 = __builtin_amdgcn_wmma_f32_16x16x32_bf16(false, af, false, b1, (short)0, acc1, false, false);
    acc2 = __builtin_amdgcn_wmma_f32_16x16x32_bf16(false, af, false, b2, (short)0, acc2, false, false);
    acc3 = __builtin_amdgcn_wmma_f32_16x16x32_bf16(false, af, false, b3, (short)0, acc3, false, false);
  }

  float rel_s = 0.f, ta_s = 0.f;
  v8f accs[4] = {acc0, acc1, acc2, acc3};
#pragma unroll
  for (int ct = 0; ct < 4; ++ct) {
    int col = ct * 16 + n;
    if (col >= 52) continue;
#pragma unroll
    for (int r = 0; r < 8; ++r) {
      int e = ebase + r + 8 * hl;
      if (e >= E) continue;
      float z = accs[ct][r];
      if (col < 19) {
        z += b_rel[col];
        rel_s += bce_term(z, edge_rel[(size_t)e * 19 + col]);
      } else {
        int c2 = col - 19;
        z += b_ta[c2];
        ta_s += bce_term(z, edge_ta[(size_t)e * 33 + c2]);
      }
    }
  }
  block_reduce_add2(rel_s, ta_s, &acc[2], &acc[3], red);
}

__global__ void k_final(const float* __restrict__ acc, float* __restrict__ out,
                        int N, int E) {
  out[0] = acc[0] / ((float)N * 4.f) + acc[1] / ((float)N * 9.f) +
           acc[2] / ((float)E * 19.f) + acc[3] / ((float)E * 33.f);
}

// ---------------------------------------------------------------------------
extern "C" void kernel_launch(void* const* d_in, const int* in_sizes, int n_in,
                              void* d_out, int out_size, void* d_ws, size_t ws_size,
                              hipStream_t stream) {
  const int D = in_sizes[7];        // b_node
  const int N = in_sizes[2] / 4;    // node_att [N,4]
  const int E = in_sizes[4] / 19;   // edge_rel [E,19]
  const int K2 = 2 * D;

  const float* x        = (const float*)d_in[0];
  const int*   ei       = (const int*)d_in[1];
  const int*   src      = ei;
  const int*   trg      = ei + E;
  const float* node_att = (const float*)d_in[2];
  const float* node_ia  = (const float*)d_in[3];
  const float* edge_rel = (const float*)d_in[4];
  const float* edge_ta  = (const float*)d_in[5];
  const float* W_node   = (const float*)d_in[6];
  const float* b_node   = (const float*)d_in[7];
  const float* W_g1     = (const float*)d_in[8];
  const float* b_g1     = (const float*)d_in[9];
  const float* W_g2     = (const float*)d_in[10];
  const float* b_g2     = (const float*)d_in[11];
  const float* W_att    = (const float*)d_in[12];
  const float* b_att    = (const float*)d_in[13];
  const float* W_ia     = (const float*)d_in[14];
  const float* b_ia     = (const float*)d_in[15];
  const float* W_rel    = (const float*)d_in[16];
  const float* b_rel    = (const float*)d_in[17];
  const float* W_ta     = (const float*)d_in[18];
  const float* b_ta     = (const float*)d_in[19];

  const size_t ND = (size_t)N * D;
  auto align256 = [](size_t x) { return (x + 255) & ~(size_t)255; };
  char* base = (char*)d_ws;
  size_t cur = 0;
  float* acc = (float*)(base + cur); cur = align256(cur + 16 * sizeof(float));
  int*   deg = (int*)(base + cur);   cur = align256(cur + (size_t)N * sizeof(int));
  float* dis = (float*)(base + cur); cur = align256(cur + (size_t)N * sizeof(float));
  float* agg = (float*)(base + cur); cur = align256(cur + ND * sizeof(float));
  bf16_t* abf0 = (bf16_t*)(base + cur); cur = align256(cur + ND * sizeof(bf16_t));
  bf16_t* abf1 = (bf16_t*)(base + cur); cur = align256(cur + ND * sizeof(bf16_t));
  bf16_t* mbf  = (bf16_t*)(base + cur); cur = align256(cur + ND * sizeof(bf16_t));
  bf16_t* wt_node = (bf16_t*)(base + cur); cur = align256(cur + (size_t)D * D * sizeof(bf16_t));
  bf16_t* wt_g1   = (bf16_t*)(base + cur); cur = align256(cur + (size_t)D * D * sizeof(bf16_t));
  bf16_t* wt_g2   = (bf16_t*)(base + cur); cur = align256(cur + (size_t)D * D * sizeof(bf16_t));
  bf16_t* wcat_t  = (bf16_t*)(base + cur); cur = align256(cur + (size_t)64 * K2 * sizeof(bf16_t));

  const unsigned nd4b = (unsigned)((ND / 4 + 255) / 256);
  const unsigned ddb  = (unsigned)(((size_t)D * D + 255) / 256);
  dim3 ggrid((N + 15) / 16, D / 128);

  k_zero_acc<<<1, 16, 0, stream>>>(acc);
  k_init_deg<<<(N + 255) / 256, 256, 0, stream>>>(deg, N);
  k_count_deg<<<(E + 255) / 256, 256, 0, stream>>>(trg, deg, E);
  k_dis<<<(N + 255) / 256, 256, 0, stream>>>(deg, dis, N);

  // bf16 weight prep (transposed, K-contiguous)
  k_transpose_w<<<ddb, 256, 0, stream>>>(W_node, wt_node, D, D);
  k_transpose_w<<<ddb, 256, 0, stream>>>(W_g1, wt_g1, D, D);
  k_transpose_w<<<ddb, 256, 0, stream>>>(W_g2, wt_g2, D, D);
  k_build_wcat<<<(unsigned)((64 * (size_t)K2 + 255) / 256), 256, 0, stream>>>(W_rel, W_ta, wcat_t, K2);

  // x -> bf16
  k_f32_to_bf16<<<nd4b, 256, 0, stream>>>(x, abf0, ND / 4);

  // h = relu(x @ W_node + b_node)  -> abf1
  k_gemm_wmma<<<ggrid, 256, 0, stream>>>(abf0, wt_node, b_node, abf1, N, D, D, 1);

  // GCN layer 1
  k_gemm_wmma<<<ggrid, 256, 0, stream>>>(abf1, wt_g1, nullptr, mbf, N, D, D, 0);
  k_scale_init<<<nd4b, 256, 0, stream>>>(mbf, dis, agg, N, D);
  k_scatter<<<E, D / 4, 0, stream>>>(mbf, src, trg, dis, agg, E, D);
  k_bias_relu<<<nd4b, 256, 0, stream>>>(agg, b_g1, abf0, N, D);

  // GCN layer 2 -> embed in abf1
  k_gemm_wmma<<<ggrid, 256, 0, stream>>>(abf0, wt_g2, nullptr, mbf, N, D, D, 0);
  k_scale_init<<<nd4b, 256, 0, stream>>>(mbf, dis, agg, N, D);
  k_scatter<<<E, D / 4, 0, stream>>>(mbf, src, trg, dis, agg, E, D);
  k_bias_relu<<<nd4b, 256, 0, stream>>>(agg, b_g2, abf1, N, D);

  // heads + losses
  k_node_heads<<<(N * 13 + 255) / 256, 256, 0, stream>>>(
      abf1, W_att, b_att, W_ia, b_ia, node_att, node_ia, acc, N, D);
  const int tiles = (E + 15) / 16;
  k_edge_heads<<<(tiles + 7) / 8, 256, 0, stream>>>(
      abf1, src, trg, wcat_t, b_rel, b_ta, edge_rel, edge_ta, acc, E, D);

  k_final<<<1, 1, 0, stream>>>(acc, (float*)d_out, N, E);
}
